// AffineTriangleAttention_91345364451473
// MI455X (gfx1250) — matrix operand order repaired and checked by reading, not compiled, not run on hardware
//
#include <hip/hip_runtime.h>

typedef unsigned short u16;
typedef unsigned int   u32;
typedef __attribute__((ext_vector_type(16))) __bf16 v16bf;
typedef __attribute__((ext_vector_type(8)))  float  v8f;
typedef __attribute__((ext_vector_type(8)))  u32    v8u;

#define N_RES    320
#define C_Z      128
#define NUM_HEAD 4
#define HEAD_DIM 32
#define M_ROWS   (N_RES * N_RES)   // 102400 rows of [C_Z]

// ---------------- helpers ----------------

// Native f32 -> bf16 conversion (compiler picks hw cvt / pack ops).
static __device__ __forceinline__ u16 f2bf(float f) {
  __bf16 b = (__bf16)f;
  return __builtin_bit_cast(u16, b);
}
// Pack two f32 -> packed bf16x2 dword.
static __device__ __forceinline__ u32 pack2bf(float lo, float hi) {
  u16 l = f2bf(lo), h = f2bf(hi);
  return (u32)l | ((u32)h << 16);
}
static __device__ __forceinline__ float bf2f(u16 h) {
  return __uint_as_float(((u32)h) << 16);
}

static __device__ __forceinline__ v8f wmma_bf16(v16bf a, v16bf b, v8f c) {
  // D(16x16 f32) = A(16x32 bf16) * B(32x16 bf16) + C
  return __builtin_amdgcn_wmma_f32_16x16x32_bf16(
      /*neg_a=*/false, a, /*neg_b=*/false, b,
      /*c_mod=*/(short)0, c, /*reuse_a=*/false, /*reuse_b=*/false);
}

// A-fragment loader: 16x32 bf16 tile, row-major rows at base + m*stride + kcol.
// ISA layout: lanes0-15 row M=lane, VGPR i = K{2i,2i+1}+half*8, VGPR 4+i = +16.
static __device__ __forceinline__ v16bf load_a_frag(const u16* base, int stride, int kcol) {
  const int lane = threadIdx.x & 31;
  const int half = lane >> 4, m = lane & 15;
  const u32* rp = (const u32*)(base + m * stride + kcol);
  v8u a;
#pragma unroll
  for (int i = 0; i < 4; i++) {
    a[i]     = rp[half * 4 + i];       // K = half*8 + {0..7}
    a[4 + i] = rp[8 + half * 4 + i];   // K = 16 + half*8 + {0..7}
  }
  return __builtin_bit_cast(v16bf, a);
}

// B-fragment loader from Bt (N x K row-major, i.e. B transposed):
// lane n holds K = kcol + half*8 + {0..7} and kcol + 16 + half*8 + {0..7} of column n.
static __device__ __forceinline__ v16bf load_bt_frag(const u16* bt, int stride, int kcol) {
  const int lane = threadIdx.x & 31;
  const int half = lane >> 4, n = lane & 15;
  const u32* rp = (const u32*)(bt + n * stride + kcol);
  v8u b;
#pragma unroll
  for (int i = 0; i < 4; i++) { b[i] = rp[half * 4 + i]; b[4 + i] = rp[8 + half * 4 + i]; }
  return __builtin_bit_cast(v16bf, b);
}

// ---------------- kernel 1: weight transpose + bf16 convert ----------------
// Wt[n][k] = W[k][n] (* scale for query). 5 matrices of 128x128.
__global__ __launch_bounds__(256) void prep_weights_kernel(
    const float* __restrict__ qw, const float* __restrict__ kw,
    const float* __restrict__ vw, const float* __restrict__ gw,
    const float* __restrict__ ow,
    u16* __restrict__ wq, u16* __restrict__ wk, u16* __restrict__ wv,
    u16* __restrict__ wg, u16* __restrict__ wo) {
  int idx = blockIdx.x * 256 + threadIdx.x;       // 0 .. 5*16384-1
  int mat = idx >> 14;
  int e   = idx & 16383;
  int n   = e >> 7, k = e & 127;
  const float* src; u16* dst; float s = 1.0f;
  if      (mat == 0) { src = qw; dst = wq; s = 0.17677669529663687f; } // 1/sqrt(32)
  else if (mat == 1) { src = kw; dst = wk; }
  else if (mat == 2) { src = vw; dst = wv; }
  else if (mat == 3) { src = gw; dst = wg; }
  else               { src = ow; dst = wo; }
  dst[n * C_Z + k] = f2bf(src[k * C_Z + n] * s);
}

// ---------------- kernel 2: fused LN + projections + nonbatched bias ----------------
// Block = 256 thr = 8 waves; wave owns 16 rows. Grid = M_ROWS/128 = 800.
__global__ __launch_bounds__(256) void ln_proj_kernel(
    const float* __restrict__ pair_act, const float* __restrict__ affine_act,
    const float* __restrict__ p_sc, const float* __restrict__ p_of,
    const float* __restrict__ a_sc, const float* __restrict__ a_of,
    const float* __restrict__ f2d,
    const u16* __restrict__ wq, const u16* __restrict__ wk,
    const u16* __restrict__ wv, const u16* __restrict__ wg,
    u16* __restrict__ qbuf, u16* __restrict__ kbuf, u16* __restrict__ vtbuf,
    u16* __restrict__ gbuf, float* __restrict__ nbbuf) {
  __shared__ u16 lds[8 * 16 * C_Z];
  const int wave = threadIdx.x >> 5;
  const int lane = threadIdx.x & 31;
  const int half = lane >> 4;
  const int lrow = lane & 15;
  const int tile0 = blockIdx.x * 128 + wave * 16;  // first global row of tile
  const int r = tile0 + lrow;                      // this lane's LN row
  u16* atile = lds + wave * (16 * C_Z);
  u32* atile32 = (u32*)atile;

  // ---- LayerNorm(pair_act row) -> bf16 A tile in LDS ----
  {
    const float4* xp = (const float4*)(pair_act + (size_t)r * C_Z + half * 64);
    float4 xv[16];
    float sum = 0.f, ss = 0.f;
#pragma unroll
    for (int i = 0; i < 16; i++) {
      float4 v = xp[i]; xv[i] = v;
      sum += v.x + v.y + v.z + v.w;
      ss  += v.x * v.x + v.y * v.y + v.z * v.z + v.w * v.w;
    }
    sum += __shfl_xor(sum, 16, 32);
    ss  += __shfl_xor(ss, 16, 32);
    float mean = sum * (1.f / C_Z);
    float rstd = rsqrtf(ss * (1.f / C_Z) - mean * mean + 1e-5f);
#pragma unroll
    for (int i = 0; i < 16; i++) {
      int c = half * 64 + i * 4;
      float n0 = (xv[i].x - mean) * rstd * p_sc[c + 0] + p_of[c + 0];
      float n1 = (xv[i].y - mean) * rstd * p_sc[c + 1] + p_of[c + 1];
      float n2 = (xv[i].z - mean) * rstd * p_sc[c + 2] + p_of[c + 2];
      float n3 = (xv[i].w - mean) * rstd * p_sc[c + 3] + p_of[c + 3];
      atile32[lrow * 64 + half * 32 + i * 2 + 0] = pack2bf(n0, n1);
      atile32[lrow * 64 + half * 32 + i * 2 + 1] = pack2bf(n2, n3);
    }
  }

  // ---- LayerNorm(affine_act row) . feat_2d_weights -> nonbatched bias [h][r] ----
  {
    const float4* xp = (const float4*)(affine_act + (size_t)r * C_Z + half * 64);
    float4 xv[16];
    float sum = 0.f, ss = 0.f;
#pragma unroll
    for (int i = 0; i < 16; i++) {
      float4 v = xp[i]; xv[i] = v;
      sum += v.x + v.y + v.z + v.w;
      ss  += v.x * v.x + v.y * v.y + v.z * v.z + v.w * v.w;
    }
    sum += __shfl_xor(sum, 16, 32);
    ss  += __shfl_xor(ss, 16, 32);
    float mean = sum * (1.f / C_Z);
    float rstd = rsqrtf(ss * (1.f / C_Z) - mean * mean + 1e-5f);
    float d0 = 0.f, d1 = 0.f, d2 = 0.f, d3 = 0.f;
    const float4* f4 = (const float4*)f2d;   // [c][4 heads]
#pragma unroll
    for (int i = 0; i < 16; i++) {
      const float* xe = (const float*)&xv[i];
#pragma unroll
      for (int t = 0; t < 4; t++) {
        int c = half * 64 + i * 4 + t;
        float val = (xe[t] - mean) * rstd * a_sc[c] + a_of[c];
        float4 fw = f4[c];
        d0 += val * fw.x; d1 += val * fw.y; d2 += val * fw.z; d3 += val * fw.w;
      }
    }
    d0 += __shfl_xor(d0, 16, 32); d1 += __shfl_xor(d1, 16, 32);
    d2 += __shfl_xor(d2, 16, 32); d3 += __shfl_xor(d3, 16, 32);
    if (half == 0) {
      nbbuf[0 * M_ROWS + r] = d0;
      nbbuf[1 * M_ROWS + r] = d1;
      nbbuf[2 * M_ROWS + r] = d2;
      nbbuf[3 * M_ROWS + r] = d3;
    }
  }

  // ---- 4 projections: [16 x 128] x [128 x 128] each, via WMMA ----
  v16bf afrag[4];
#pragma unroll
  for (int kk = 0; kk < 4; kk++) afrag[kk] = load_a_frag(atile, C_Z, kk * 32);

  const int bb = tile0 / N_RES;   // residue-i (batch of attention)
  const int q0 = tile0 % N_RES;   // first q row within this b
  const int ln = lrow;            // D-fragment N = lane&15

  // query -> qbuf[(b,h)][row][c], scale already folded into wq
#pragma unroll
  for (int nt = 0; nt < 8; nt++) {
    v8f acc = {0.f,0.f,0.f,0.f,0.f,0.f,0.f,0.f};
#pragma unroll
    for (int kk = 0; kk < 4; kk++)
      acc = wmma_bf16(afrag[kk], load_bt_frag(wq + nt * 16 * C_Z, C_Z, kk * 32), acc);
    int co = nt * 16 + ln, h = co >> 5, c = co & 31;
#pragma unroll
    for (int j = 0; j < 8; j++)
      qbuf[((size_t)(bb * NUM_HEAD + h) * N_RES + (q0 + half * 8 + j)) * HEAD_DIM + c] = f2bf(acc[j]);
  }
  // key -> kbuf[(b,h)][row][c]
#pragma unroll
  for (int nt = 0; nt < 8; nt++) {
    v8f acc = {0.f,0.f,0.f,0.f,0.f,0.f,0.f,0.f};
#pragma unroll
    for (int kk = 0; kk < 4; kk++)
      acc = wmma_bf16(afrag[kk], load_bt_frag(wk + nt * 16 * C_Z, C_Z, kk * 32), acc);
    int co = nt * 16 + ln, h = co >> 5, c = co & 31;
#pragma unroll
    for (int j = 0; j < 8; j++)
      kbuf[((size_t)(bb * NUM_HEAD + h) * N_RES + (q0 + half * 8 + j)) * HEAD_DIM + c] = f2bf(acc[j]);
  }
  // value -> TRANSPOSED vtbuf[(b,h)][c][row] so PV B-fragments are contiguous
#pragma unroll
  for (int nt = 0; nt < 8; nt++) {
    v8f acc = {0.f,0.f,0.f,0.f,0.f,0.f,0.f,0.f};
#pragma unroll
    for (int kk = 0; kk < 4; kk++)
      acc = wmma_bf16(afrag[kk], load_bt_frag(wv + nt * 16 * C_Z, C_Z, kk * 32), acc);
    int co = nt * 16 + ln, h = co >> 5, c = co & 31;
#pragma unroll
    for (int j = 0; j < 8; j++)
      vtbuf[((size_t)(bb * NUM_HEAD + h) * HEAD_DIM + c) * N_RES + (q0 + half * 8 + j)] = f2bf(acc[j]);
  }
  // gate -> sigmoid -> gbuf[r][co]
#pragma unroll
  for (int nt = 0; nt < 8; nt++) {
    v8f acc = {0.f,0.f,0.f,0.f,0.f,0.f,0.f,0.f};
#pragma unroll
    for (int kk = 0; kk < 4; kk++)
      acc = wmma_bf16(afrag[kk], load_bt_frag(wg + nt * 16 * C_Z, C_Z, kk * 32), acc);
    int co = nt * 16 + ln;
#pragma unroll
    for (int j = 0; j < 8; j++) {
      float g = 1.f / (1.f + __expf(-acc[j]));
      gbuf[(size_t)(tile0 + half * 8 + j) * C_Z + co] = f2bf(g);
    }
  }
}

// ---------------- kernel 3: flash attention per (b, h, 16-q-row tile) ----------------
// Block = 128 thr = 4 waves; grid (5, NUM_HEAD, N_RES); wave w -> qtile = bx*4+w.
__global__ __launch_bounds__(128) void attn_kernel(
    const u16* __restrict__ qbuf, const u16* __restrict__ kbuf,
    const u16* __restrict__ vtbuf, const float* __restrict__ nbbuf,
    const float* __restrict__ mask, u16* __restrict__ wabuf) {
  __shared__ u16 pstage[4 * 16 * 32];
  const int wave = threadIdx.x >> 5;
  const int lane = threadIdx.x & 31;
  const int half = lane >> 4;
  const int ln   = lane & 15;
  const int b = blockIdx.z;
  const int h = blockIdx.y;
  const int q0 = (blockIdx.x * 4 + wave) * 16;
  const u16* qh = qbuf + (size_t)(b * NUM_HEAD + h) * N_RES * HEAD_DIM;
  const u16* kh = kbuf + (size_t)(b * NUM_HEAD + h) * N_RES * HEAD_DIM;
  const u16* vh = vtbuf + (size_t)(b * NUM_HEAD + h) * HEAD_DIM * N_RES; // [c][row]
  const float* nbh = nbbuf + (size_t)h * M_ROWS;                          // [q][k]
  u16* stage = pstage + wave * (16 * 32);

  const v16bf aq = load_a_frag(qh + q0 * HEAD_DIM, HEAD_DIM, 0);  // q tile 16x32
  v8f o0 = {0.f,0.f,0.f,0.f,0.f,0.f,0.f,0.f};
  v8f o1 = {0.f,0.f,0.f,0.f,0.f,0.f,0.f,0.f};
  float ms[8], ls[8];
#pragma unroll
  for (int j = 0; j < 8; j++) { ms[j] = -1e30f; ls[j] = 0.f; }

  for (int k0 = 0; k0 < N_RES; k0 += 32) {
    // Prefetch next key/value tiles into cache (global_prefetch_b8).
    if (k0 + 32 < N_RES) {
      __builtin_prefetch(kh + (size_t)(k0 + 32 + (lane >> 1)) * HEAD_DIM, 0, 3);
      __builtin_prefetch(vh + (size_t)lane * N_RES + k0 + 32, 0, 3);
    }

    // S = q * k^T for 32 keys (two 16-wide N chunks), K = head_dim = 32
    v16bf bk0 = load_bt_frag(kh + (size_t)k0 * HEAD_DIM, HEAD_DIM, 0);
    v16bf bk1 = load_bt_frag(kh + (size_t)(k0 + 16) * HEAD_DIM, HEAD_DIM, 0);
    v8f zero = {0.f,0.f,0.f,0.f,0.f,0.f,0.f,0.f};
    v8f s0 = wmma_bf16(aq, bk0, zero);
    v8f s1 = wmma_bf16(aq, bk1, zero);

    float mb0 = 1e9f * (mask[b * N_RES + k0 + ln] - 1.f);
    float mb1 = 1e9f * (mask[b * N_RES + k0 + 16 + ln] - 1.f);
    const float* nbp = nbh + (size_t)(q0 + half * 8) * N_RES + k0 + ln;

#pragma unroll
    for (int j = 0; j < 8; j++) {
      float s0j = s0[j] + mb0 + nbp[j * N_RES];
      float s1j = s1[j] + mb1 + nbp[j * N_RES + 16];
      // online softmax; D-layout: this slot is row = half*8 + j, key = ln(+16)
      float mx = fmaxf(s0j, s1j);
#pragma unroll
      for (int d = 1; d < 16; d <<= 1) mx = fmaxf(mx, __shfl_xor(mx, d, 32));
      float mnew  = fmaxf(ms[j], mx);
      float alpha = __expf(ms[j] - mnew);
      float p0 = __expf(s0j - mnew);
      float p1 = __expf(s1j - mnew);
      float rs = p0 + p1;
#pragma unroll
      for (int d = 1; d < 16; d <<= 1) rs += __shfl_xor(rs, d, 32);
      ls[j] = ls[j] * alpha + rs;
      ms[j] = mnew;
      o0[j] *= alpha;
      o1[j] *= alpha;
      const int row = half * 8 + j;
      stage[row * 32 + ln]      = f2bf(p0);
      stage[row * 32 + 16 + ln] = f2bf(p1);
    }
    __builtin_amdgcn_wave_barrier();  // keep DS store->load ordering tight (intra-wave)

    // O += P(16x32) * V(32xHEAD_DIM); V transposed so Bt rows are contiguous keys
    v16bf ap  = load_a_frag(stage, 32, 0);
    v16bf bv0 = load_bt_frag(vh, N_RES, k0);
    v16bf bv1 = load_bt_frag(vh + 16 * N_RES, N_RES, k0);
    o0 = wmma_bf16(ap, bv0, o0);
    o1 = wmma_bf16(ap, bv1, o1);
  }

  // epilogue: divide by softmax sum, store weighted_avg bf16 [r][h*32 + c]
#pragma unroll
  for (int j = 0; j < 8; j++) {
    const int row = half * 8 + j;
    const float inv = 1.f / ls[j];
    size_t base = ((size_t)(b * N_RES + q0 + row)) * C_Z + h * HEAD_DIM;
    wabuf[base + ln]      = f2bf(o0[j] * inv);
    wabuf[base + 16 + ln] = f2bf(o1[j] * inv);
  }
}

// ---------------- kernel 4: gate * weighted_avg, output projection ----------------
__global__ __launch_bounds__(256) void gate_out_kernel(
    const u16* __restrict__ wabuf, const u16* __restrict__ gbuf,
    const u16* __restrict__ wo, float* __restrict__ out) {
  __shared__ u16 lds[8 * 16 * C_Z];
  const int wave = threadIdx.x >> 5;
  const int lane = threadIdx.x & 31;
  const int half = lane >> 4;
  const int lrow = lane & 15;
  const int tile0 = blockIdx.x * 128 + wave * 16;
  const int r = tile0 + lrow;
  u16* atile = lds + wave * (16 * C_Z);
  u32* atile32 = (u32*)atile;

  const u32* wrow = (const u32*)(wabuf + (size_t)r * C_Z + half * 64);
  const u32* grow = (const u32*)(gbuf + (size_t)r * C_Z + half * 64);
#pragma unroll
  for (int i = 0; i < 32; i++) {
    u32 w = wrow[i], g = grow[i];
    float a0 = bf2f((u16)w) * bf2f((u16)g);
    float a1 = bf2f((u16)(w >> 16)) * bf2f((u16)(g >> 16));
    atile32[lrow * 64 + half * 32 + i] = pack2bf(a0, a1);
  }

  v16bf afrag[4];
#pragma unroll
  for (int kk = 0; kk < 4; kk++) afrag[kk] = load_a_frag(atile, C_Z, kk * 32);

#pragma unroll
  for (int nt = 0; nt < 8; nt++) {
    v8f acc = {0.f,0.f,0.f,0.f,0.f,0.f,0.f,0.f};
#pragma unroll
    for (int kk = 0; kk < 4; kk++)
      acc = wmma_bf16(afrag[kk], load_bt_frag(wo + nt * 16 * C_Z, C_Z, kk * 32), acc);
    int co = nt * 16 + (lane & 15);
#pragma unroll
    for (int j = 0; j < 8; j++)
      out[(size_t)(tile0 + half * 8 + j) * C_Z + co] = acc[j];
  }
}

// ---------------- host launcher ----------------
extern "C" void kernel_launch(void* const* d_in, const int* in_sizes, int n_in,
                              void* d_out, int out_size, void* d_ws, size_t ws_size,
                              hipStream_t stream) {
  (void)in_sizes; (void)n_in; (void)out_size; (void)ws_size;
  const float* pair_act     = (const float*)d_in[0];
  const float* affine_act   = (const float*)d_in[1];
  const float* pair_mask    = (const float*)d_in[2];
  const float* pair_sc      = (const float*)d_in[3];
  const float* pair_of      = (const float*)d_in[4];
  const float* aff_sc       = (const float*)d_in[5];
  const float* aff_of       = (const float*)d_in[6];
  const float* f2d          = (const float*)d_in[7];
  const float* query_w      = (const float*)d_in[8];
  const float* key_w        = (const float*)d_in[9];
  const float* value_w      = (const float*)d_in[10];
  const float* gating_w     = (const float*)d_in[11];
  const float* output_w     = (const float*)d_in[12];
  float* out = (float*)d_out;

  char* p = (char*)d_ws;
  auto take = [&](size_t bytes) -> void* {
    void* q = (void*)p;
    p += (bytes + 255) & ~(size_t)255;
    return q;
  };
  u16* wq = (u16*)take((size_t)C_Z * C_Z * 2);
  u16* wk = (u16*)take((size_t)C_Z * C_Z * 2);
  u16* wv = (u16*)take((size_t)C_Z * C_Z * 2);
  u16* wg = (u16*)take((size_t)C_Z * C_Z * 2);
  u16* wo = (u16*)take((size_t)C_Z * C_Z * 2);
  const size_t headed = (size_t)N_RES * NUM_HEAD * N_RES * HEAD_DIM * 2;  // 26.2 MB
  u16* qbuf  = (u16*)take(headed);
  u16* kbuf  = (u16*)take(headed);
  u16* vtbuf = (u16*)take(headed);
  u16* gbuf  = (u16*)take((size_t)M_ROWS * C_Z * 2);
  u16* wabuf = (u16*)take((size_t)M_ROWS * C_Z * 2);
  float* nbbuf = (float*)take((size_t)NUM_HEAD * M_ROWS * 4);

  prep_weights_kernel<<<(5 * C_Z * C_Z) / 256, 256, 0, stream>>>(
      query_w, key_w, value_w, gating_w, output_w, wq, wk, wv, wg, wo);

  ln_proj_kernel<<<M_ROWS / 128, 256, 0, stream>>>(
      pair_act, affine_act, pair_sc, pair_of, aff_sc, aff_of, f2d,
      wq, wk, wv, wg, qbuf, kbuf, vtbuf, gbuf, nbbuf);

  dim3 gb(N_RES / 64, NUM_HEAD, N_RES);  // 5 x 4 x 320, 4 waves/block
  attn_kernel<<<gb, 128, 0, stream>>>(qbuf, kbuf, vtbuf, nbbuf, pair_mask, wabuf);

  gate_out_kernel<<<M_ROWS / 128, 256, 0, stream>>>(wabuf, gbuf, wo, out);
}
